// GINERegressor_61950608277982
// MI455X (gfx1250) — matrix operand order repaired and checked by reading, not compiled, not run on hardware
//
#include <hip/hip_runtime.h>
#include <hip/hip_bf16.h>

typedef _Float16 h16;
typedef __attribute__((ext_vector_type(16))) _Float16 v16h;
typedef __attribute__((ext_vector_type(8)))  _Float16 v8h_t;
typedef __attribute__((ext_vector_type(8)))  float    v8f;

#define BB 8
#define NN 128
#define F_IN 64
#define HH 256
#define LL 8
#define AST 264   // padded LDS row stride in halves (264 mod 64-bank pattern -> conflict free)
#define EST 40    // edge tile stride in halves (32 data + 8 pad)

// ---------------------------------------------------------------------------
// LDS A-fragment load for v_wmma_f32_16x16x32_f16 (contiguous K order used by this file, not the ISA table order):
// lane<16: M=lane,     K = k0+{0..15}
// lane>=16:M=lane-16,  K = k0+{16..31}
// -> two contiguous 8-half (16B) runs per lane.
__device__ inline v16h load_a_frag(const h16* base, int row, int stride, int bk) {
    const v8h_t lo = *(const v8h_t*)(base + row * stride + bk);
    const v8h_t hi = *(const v8h_t*)(base + row * stride + bk + 8);
    v16h r;
#pragma unroll
    for (int e = 0; e < 8; ++e) { r[e] = lo[e]; r[e + 8] = hi[e]; }
    return r;
}

// ---------------------------------------------------------------------------
// Kernel 1: weight precondition. W1t/W2t layout [L][N][K] f16 (transposed) so a
// B-fragment (lane = column n, K = k0 + 16*(lane>>4) + 0..15) is one contiguous
// 32B load. edge_W (16xH) -> eWt [N][32] f16 zero-padded to K=32.
__global__ __launch_bounds__(256) void gine_prep(const float* __restrict__ W1,
                                                 const float* __restrict__ W2,
                                                 const float* __restrict__ eW,
                                                 h16* W1t, h16* W2t, h16* eWt) {
    const int WSZ = LL * HH * HH;  // 524288
    int id = blockIdx.x * 256 + threadIdx.x;
    if (id < WSZ) {
        int l = id >> 16, rem = id & 65535, n = rem >> 8, k = rem & 255;
        W1t[id] = (h16)W1[(l << 16) + (k << 8) + n];
    } else if (id < 2 * WSZ) {
        int id2 = id - WSZ;
        int l = id2 >> 16, rem = id2 & 65535, n = rem >> 8, k = rem & 255;
        W2t[id2] = (h16)W2[(l << 16) + (k << 8) + n];
    } else if (id < 2 * WSZ + HH * 32) {
        int e = id - 2 * WSZ;
        int n = e >> 5, k = e & 31;
        eWt[e] = (k < 16) ? (h16)eW[(k << 8) + n] : (h16)0.f;
    }
}

// ---------------------------------------------------------------------------
// Kernel 2: node embedding  h = x @ node_W + node_b   (1024 x 64) @ (64 x 256)
__global__ __launch_bounds__(256) void gine_node(const float* __restrict__ x,
                                                 const float* __restrict__ nW,
                                                 const float* __restrict__ nb,
                                                 float* hbuf) {
    __shared__ float xr[F_IN];
    int row = blockIdx.x, t = threadIdx.x;
    if (t < F_IN) xr[t] = x[row * F_IN + t];
    __syncthreads();
    float acc = nb[t];
    for (int k = 0; k < F_IN; ++k) acc += xr[k] * nW[k * HH + t];
    hbuf[(size_t)row * HH + t] = acc;
}

// ---------------------------------------------------------------------------
// Kernel 3: per-layer mask + layernorm.  h <- h*m ;  hn <- LN(h)*g+b  (f16)
__global__ __launch_bounds__(256) void gine_ln(float* hbuf, h16* hn,
                                               const int* __restrict__ mask,
                                               const float* __restrict__ ln_g,
                                               const float* __restrict__ ln_b,
                                               int layer) {
    __shared__ float red[HH];
    __shared__ float mu_s, rs_s;
    int row = blockIdx.x, t = threadIdx.x;
    float mv = (float)mask[row];
    float v = hbuf[(size_t)row * HH + t] * mv;
    hbuf[(size_t)row * HH + t] = v;
    red[t] = v;
    __syncthreads();
    for (int s = 128; s > 0; s >>= 1) { if (t < s) red[t] += red[t + s]; __syncthreads(); }
    if (t == 0) mu_s = red[0] * (1.f / HH);
    __syncthreads();
    float d = v - mu_s;
    red[t] = d * d;
    __syncthreads();
    for (int s = 128; s > 0; s >>= 1) { if (t < s) red[t] += red[t + s]; __syncthreads(); }
    if (t == 0) rs_s = rsqrtf(red[0] * (1.f / HH) + 1e-5f);
    __syncthreads();
    float hnv = d * rs_s * ln_g[layer * HH + t] + ln_b[layer * HH + t];
    hn[(size_t)row * HH + t] = (h16)hnv;
}

// ---------------------------------------------------------------------------
// Kernel 4: fused message pass for one (b,i). 8 waves, wave w owns row-tile w
// (16 j-rows). Stages:
//   A = f16( hn[b,j,:] + edge_attr[b,i,j,:] @ eWt + edge_b )   via WMMA (K=32 pad)
//   T1 = f16( relu(A @ W1t + b1) )                              via WMMA
//   mm = relu(T1 @ W2t + b2) * adj[b,i,j]; msg = column-sum(mm) via WMMA + ds_add
//   h[b,i,:] = (h[b,i,:] + msg) * m
__global__ __launch_bounds__(256) void gine_msg(float* hbuf,
                                                const h16* __restrict__ hn,
                                                const h16* __restrict__ W1t,
                                                const h16* __restrict__ W2t,
                                                const h16* __restrict__ eWt,
                                                const float* __restrict__ edge_attr,
                                                const int* __restrict__ adj,
                                                const int* __restrict__ mask,
                                                const float* __restrict__ edge_b,
                                                const float* __restrict__ b1g,
                                                const float* __restrict__ b2g,
                                                int layer) {
    extern __shared__ char smem[];
    h16*   A    = (h16*)smem;           // 128 x AST halves
    h16*   T1   = A + NN * AST;         // 128 x AST halves
    h16*   E    = T1 + NN * AST;        // 128 x EST halves
    float* adjs = (float*)(E + NN * EST);  // 128
    float* msg  = adjs + NN;               // 256

    const int b = blockIdx.x >> 7;
    const int i = blockIdx.x & (NN - 1);
    const int tid = threadIdx.x;
    const int wave = tid >> 5, lane = tid & 31;
    const int lhalf = lane >> 4, l15 = lane & 15;
    const int rowbase = wave << 4;

    // ---- stage 1: stage edge tile (f16, K padded to 32), adj row, zero msg
    const float* eap = edge_attr + ((size_t)(b * NN + i)) * NN * 16;
    for (int idx = tid; idx < NN * 16; idx += 256) {
        int j = idx >> 4, k = idx & 15;
        E[j * EST + k] = (h16)eap[idx];
        E[j * EST + 16 + k] = (h16)0.f;
    }
    if (tid < NN) adjs[tid] = (float)adj[(b * NN + i) * NN + tid];
    if (tid < HH) msg[tid] = 0.f;
    __syncthreads();

    // ---- stage 2: edge GEMM (K=32, one wmma per n-tile) + hn + edge bias -> A
    {
        v16h af = load_a_frag(E, rowbase + l15, EST, lhalf << 4);
        const h16* hnb = hn + (size_t)(b * NN) * HH;
        for (int nt = 0; nt < 16; ++nt) {
            int n = (nt << 4) + l15;
            v16h bf = *(const v16h*)(eWt + n * 32 + (lhalf << 4));
            v8f acc = {};
            acc = __builtin_amdgcn_wmma_f32_16x16x32_f16(false, af, false, bf,
                                                         (short)0, acc, false, false);
            float bias = edge_b[n];
#pragma unroll
            for (int r = 0; r < 8; ++r) {
                int j = rowbase + r + (lhalf << 3);
                float v = acc[r] + bias + (float)hnb[j * HH + n];
                A[j * AST + n] = (h16)v;
            }
        }
    }
    __syncthreads();

    // ---- stage 3: T1 = relu(A @ W1t + b1)
    {
        const h16* Wl = W1t + (size_t)layer * HH * HH;
        for (int nt = 0; nt < 16; ++nt) {
            int n = (nt << 4) + l15;
            v8f acc = {};
#pragma unroll
            for (int ks = 0; ks < 8; ++ks) {
                int k0 = ks << 5;
                v16h af = load_a_frag(A, rowbase + l15, AST, k0 + (lhalf << 4));
                v16h bf = *(const v16h*)(Wl + (size_t)n * HH + k0 + (lhalf << 4));
                acc = __builtin_amdgcn_wmma_f32_16x16x32_f16(false, af, false, bf,
                                                             (short)0, acc, false, false);
            }
            float bias = b1g[layer * HH + n];
#pragma unroll
            for (int r = 0; r < 8; ++r) {
                int j = rowbase + r + (lhalf << 3);
                float v = acc[r] + bias;
                T1[j * AST + n] = (h16)(v > 0.f ? v : 0.f);
            }
        }
    }
    __syncthreads();

    // ---- stage 4: mm = relu(T1 @ W2t + b2) * adj; column-reduce into msg
    {
        const h16* Wl = W2t + (size_t)layer * HH * HH;
        for (int nt = 0; nt < 16; ++nt) {
            int n = (nt << 4) + l15;
            v8f acc = {};
#pragma unroll
            for (int ks = 0; ks < 8; ++ks) {
                int k0 = ks << 5;
                v16h af = load_a_frag(T1, rowbase + l15, AST, k0 + (lhalf << 4));
                v16h bf = *(const v16h*)(Wl + (size_t)n * HH + k0 + (lhalf << 4));
                acc = __builtin_amdgcn_wmma_f32_16x16x32_f16(false, af, false, bf,
                                                             (short)0, acc, false, false);
            }
            float bias = b2g[layer * HH + n];
            float part = 0.f;
#pragma unroll
            for (int r = 0; r < 8; ++r) {
                int j = rowbase + r + (lhalf << 3);
                float v = acc[r] + bias;
                part += (v > 0.f ? v : 0.f) * adjs[j];
            }
            atomicAdd(&msg[n], part);
        }
    }
    __syncthreads();

    // ---- stage 5: residual + mask
    {
        float mv = (float)mask[b * NN + i];
        float* hrow = hbuf + (size_t)(b * NN + i) * HH;
        if (tid < HH) hrow[tid] = (hrow[tid] + msg[tid]) * mv;
    }
}

// ---------------------------------------------------------------------------
// Kernel 5: attention readout (tail; VALU is fine)
__global__ __launch_bounds__(256) void gine_readout(const float* __restrict__ hbuf,
                                                    const int* __restrict__ mask,
                                                    const float* __restrict__ pW,
                                                    const float* __restrict__ pb,
                                                    const float* __restrict__ aW,
                                                    const float* __restrict__ ab,
                                                    const float* __restrict__ phW,
                                                    const float* __restrict__ phb,
                                                    float* out) {
    __shared__ float sc[NN];
    __shared__ float alpha[NN];
    __shared__ float red[HH];
    __shared__ float mx_s, sum_s, cnt_s;
    int b = blockIdx.x, t = threadIdx.x;
    if (t < NN) {
        const float* hr = hbuf + (size_t)(b * NN + t) * HH;
        float s = 0.f;
        for (int n = 0; n < HH; ++n) {
            float acc = pb[n];
            for (int k = 0; k < HH; ++k) acc += hr[k] * pW[k * HH + n];
            s += tanhf(acc) * aW[n];
        }
        s += ab[0];
        if (mask[b * NN + t] == 0) s = -__builtin_inff();
        sc[t] = s;
    }
    __syncthreads();
    if (t == 0) {
        float mx = -__builtin_inff();
        for (int j = 0; j < NN; ++j) mx = sc[j] > mx ? sc[j] : mx;
        float sum = 0.f;
        if (mx != -__builtin_inff())
            for (int j = 0; j < NN; ++j) sum += __expf(sc[j] - mx);
        float cnt = 0.f;
        for (int j = 0; j < NN; ++j) cnt += (float)mask[b * NN + j];
        if (cnt < 1.f) cnt = 1.f;
        mx_s = mx; sum_s = sum; cnt_s = cnt;
    }
    __syncthreads();
    if (t < NN) {
        float a;
        if (sum_s > 0.f && mx_s != -__builtin_inff())
            a = __expf(sc[t] - mx_s) / sum_s;
        else
            a = mask[b * NN + t] ? (1.f / cnt_s) : 0.f;
        alpha[t] = a;
    }
    __syncthreads();
    float g = 0.f;
    for (int j = 0; j < NN; ++j) g += alpha[j] * hbuf[(size_t)(b * NN + j) * HH + t];
    red[t] = g * phW[t];
    __syncthreads();
    for (int s = 128; s > 0; s >>= 1) { if (t < s) red[t] += red[t + s]; __syncthreads(); }
    if (t == 0) out[b] = red[0] + phb[0];
}

// ---------------------------------------------------------------------------
static const size_t MSG_SMEM = (size_t)(2 * NN * AST + NN * EST) * sizeof(h16)
                             + (size_t)(NN + HH) * sizeof(float);  // 146944 B

extern "C" void kernel_launch(void* const* d_in, const int* in_sizes, int n_in,
                              void* d_out, int out_size, void* d_ws, size_t ws_size,
                              hipStream_t stream) {
    const float* x         = (const float*)d_in[0];
    const int*   adj       = (const int*)  d_in[1];
    const float* edge_attr = (const float*)d_in[2];
    const int*   mask      = (const int*)  d_in[3];
    const float* node_W    = (const float*)d_in[4];
    const float* node_b    = (const float*)d_in[5];
    const float* edge_W    = (const float*)d_in[6];
    const float* edge_b    = (const float*)d_in[7];
    const float* ln_g      = (const float*)d_in[8];
    const float* ln_b      = (const float*)d_in[9];
    const float* W1        = (const float*)d_in[10];
    const float* b1        = (const float*)d_in[11];
    const float* W2        = (const float*)d_in[12];
    const float* b2        = (const float*)d_in[13];
    const float* ro_pW     = (const float*)d_in[14];
    const float* ro_pb     = (const float*)d_in[15];
    const float* ro_aW     = (const float*)d_in[16];
    const float* ro_ab     = (const float*)d_in[17];
    const float* ph_W      = (const float*)d_in[18];
    const float* ph_b      = (const float*)d_in[19];
    float* out = (float*)d_out;

    // workspace carve (all offsets 32B aligned)
    char* ws = (char*)d_ws;
    float* hbuf = (float*)ws;                              // 8*128*256 f32 = 1 MB
    h16*   hn   = (h16*)(ws + (1u << 20));                 // 8*128*256 f16 = 512 KB
    h16*   W1t  = (h16*)(ws + (1u << 20) + (512u << 10));  // 1 MB
    h16*   W2t  = W1t + LL * HH * HH;                      // 1 MB
    h16*   eWt  = W2t + LL * HH * HH;                      // 16 KB

    (void)hipFuncSetAttribute(reinterpret_cast<const void*>(gine_msg),
                              hipFuncAttributeMaxDynamicSharedMemorySize,
                              (int)MSG_SMEM);

    {
        int total = 2 * LL * HH * HH + HH * 32;
        gine_prep<<<(total + 255) / 256, 256, 0, stream>>>(W1, W2, edge_W, W1t, W2t, eWt);
    }
    gine_node<<<BB * NN, 256, 0, stream>>>(x, node_W, node_b, hbuf);

    for (int layer = 0; layer < LL; ++layer) {
        gine_ln<<<BB * NN, 256, 0, stream>>>(hbuf, hn, mask, ln_g, ln_b, layer);
        gine_msg<<<BB * NN, 256, MSG_SMEM, stream>>>(hbuf, hn, W1t, W2t, eWt,
                                                     edge_attr, adj, mask,
                                                     edge_b, b1, b2, layer);
    }
    gine_readout<<<BB, 256, 0, stream>>>(hbuf, mask, ro_pW, ro_pb, ro_aW, ro_ab,
                                         ph_W, ph_b, out);
}
